// MaskTransformer_81475529605158
// MI455X (gfx1250) — matrix operand order repaired and checked by reading, not compiled, not run on hardware
//
#include <hip/hip_runtime.h>
#include <math.h>

#define Bb 8
#define Ss 1024
#define NSS 32
#define Nn 8192
#define EPSV 1e-7f

typedef __attribute__((ext_vector_type(16))) _Float16 v16h;
typedef __attribute__((ext_vector_type(8)))  _Float16 v8h;
typedef __attribute__((ext_vector_type(8)))  float    v8f;

// ---------------- workspace layout (bytes) ----------------
#define WF0_OFF   0u         // 64  x 32  f16  (K padded 8->32)
#define WF1_OFF   4096u      // 128 x 64  f16
#define WF2_OFF   20480u     // 384 x 128 f16
#define BB0_OFF   118784u    // 64  f32
#define BB1_OFF   119040u    // 128 f32
#define BB2_OFF   119552u    // 384 f32
#define FEAT_OFF  131072u    // B*S*32*8 f16 = 4 MiB

// ---------------- kernel 0: fold BN into f16 weights ----------------
__global__ void prep_layer(const float* __restrict__ W, const float* __restrict__ bv,
                           const float* __restrict__ g, const float* __restrict__ be,
                           const float* __restrict__ m, const float* __restrict__ v,
                           int ci, int co, int Kpad,
                           _Float16* __restrict__ Wf, float* __restrict__ bb) {
    int o = blockIdx.x * blockDim.x + threadIdx.x;
    if (o >= co) return;
    float s = g[o] * rsqrtf(v[o] + 1e-5f);
    bb[o] = s * (bv[o] - m[o]) + be[o];
    for (int k = 0; k < Kpad; ++k) {
        float val = (k < ci) ? W[k * co + o] * s : 0.0f;
        Wf[o * Kpad + k] = (_Float16)val;
    }
}

// ---------------- kernel 1: geometry features (1 wave = 1 (b,s)) ----------------
__device__ __forceinline__ float shfl32(float v, int src) { return __shfl(v, src, 32); }
__device__ __forceinline__ int   shfl32i(int v, int src)  { return __shfl(v, src, 32); }

__global__ void geom_kernel(const float* __restrict__ xyz, const float* __restrict__ centers,
                            const float* __restrict__ cns, const float* __restrict__ pts,
                            const float* __restrict__ nori, const int* __restrict__ idx,
                            _Float16* __restrict__ featG) {
    __shared__ float sds[4][32];
    __shared__ int   sis[4][32];
    int wave = threadIdx.x >> 5;
    int lane = threadIdx.x & 31;
    int bs   = blockIdx.x * 4 + wave;          // grid = B*S/4 exactly
    int b    = bs / Ss;

    // broadcast per-patch vectors
    float cnx = cns[bs*3+0], cny = cns[bs*3+1], cnz = cns[bs*3+2];
    float cx  = centers[bs*3+0], cy = centers[bs*3+1], cz = centers[bs*3+2];

    // per-neighbor point
    const float* p = xyz + ((size_t)bs * NSS + lane) * 3;
    float px = p[0], py = p[1], pz = p[2];

    float dist = px*cnx + py*cny + pz*cnz;
    float qx = px - dist*cnx, qy = py - dist*cny, qz = pz - dist*cnz;
    float plen = sqrtf(qx*qx + qy*qy + qz*qz);
    float ux = 0.f, uy = 0.f, uz = 0.f;
    if (plen > 0.f) { ux = qx/plen; uy = qy/plen; uz = qz/plen; }

    // argmax(plen) over wave (first index on ties)
    float bv2 = plen; int bi = lane;
    #pragma unroll
    for (int off = 16; off >= 1; off >>= 1) {
        float ov = __shfl_xor(bv2, off, 32);
        int   oi = __shfl_xor(bi,  off, 32);
        if (ov > bv2 || (ov == bv2 && oi < bi)) { bv2 = ov; bi = oi; }
    }
    float rx = shfl32(ux, bi), ry = shfl32(uy, bi), rz = shfl32(uz, bi);

    float dotv = ux*rx + uy*ry + uz*rz;
    // cross(punit, vref) . cn
    float crx = uy*rz - uz*ry, cry = uz*rx - ux*rz, crz = ux*ry - uy*rx;
    float sdot = crx*cnx + cry*cny + crz*cnz;
    float sg = (sdot > 0.f) ? 1.f : ((sdot < 0.f) ? -1.f : 0.f);
    if (lane == 0) sg = 1.f;
    float dotsAdj = sg * dotv - (1.f - sg);

    // stable descending rank
    int rank = 0;
    #pragma unroll
    for (int j = 0; j < 32; ++j) {
        float dj = shfl32(dotsAdj, j);
        if (dj > dotsAdj || (dj == dotsAdj && j < lane)) rank++;
    }

    int myidx = idx[(size_t)bs * NSS + lane];
    sds[wave][rank] = dotsAdj;
    sis[wave][rank] = myidx;
    __syncthreads();
    float ds   = sds[wave][lane];   // sorted position = lane
    int   idxo = sis[wave][lane];
    __syncthreads();

    // gather point + normal
    const float* gp = pts  + ((size_t)b * Nn + idxo) * 3;
    const float* gn = nori + ((size_t)b * Nn + idxo) * 3;
    float lx = gp[0] - cx, ly = gp[1] - cy, lz = gp[2] - cz;
    float nx = gn[0], ny = gn[1], nz = gn[2];

    float glen = sqrtf(lx*lx + ly*ly + lz*lz);
    float gx = 0.f, gy = 0.f, gz = 0.f;
    if (glen > 0.f) { gx = lx/glen; gy = ly/glen; gz = lz/glen; }

    float a0 = gx*cnx + gy*cny + gz*cnz;
    float a1 = gx*nx + gy*ny + gz*nz;
    float dn = nx*cnx + ny*cny + nz*cnz;
    dn = fminf(fmaxf(dn, -1.f + EPSV), 1.f - EPSV);
    float an = acosf(dn);

    // ring (roll by 1 in sorted order)
    int pp = (lane + 31) & 31;
    float plx = shfl32(lx, pp), ply = shfl32(ly, pp), plz = shfl32(lz, pp);
    float pnx = shfl32(nx, pp), pny = shfl32(ny, pp), pnz = shfl32(nz, pp);
    float dsp = shfl32(ds, pp);

    float ex = lx - plx, ey = ly - ply, ez = lz - plz;
    float elen = sqrtf(ex*ex + ey*ey + ez*ez);
    float ix = 0.f, iy = 0.f, iz = 0.f;
    if (elen > 0.f) { ix = ex/elen; iy = ey/elen; iz = ez/elen; }

    float ia0 = ix*nx + iy*ny + iz*nz;
    float ia1 = ix*pnx + iy*pny + iz*pnz;
    float dnn = nx*pnx + ny*pny + nz*pnz;
    dnn = fminf(fmaxf(dnn, -1.f + EPSV), 1.f - EPSV);
    float ia2 = acosf(dnn);

    float paf = (lane == 0) ? (-3.f - dsp) : (ds - dsp);

    _Float16* f = featG + ((size_t)bs * NSS + lane) * 8;
    f[0] = (_Float16)glen; f[1] = (_Float16)paf; f[2] = (_Float16)a0;  f[3] = (_Float16)a1;
    f[4] = (_Float16)an;   f[5] = (_Float16)ia0; f[6] = (_Float16)ia1; f[7] = (_Float16)ia2;
}

// ---------------- WMMA fragment helpers (ISA-documented layouts) ----------------
__device__ __forceinline__ v16h load_a_frag(const _Float16* act, int stride, int mtile,
                                            int kc, int lane) {
    int half = lane >> 4;
    int m    = (lane & 15) + mtile * 16;
    const _Float16* base = act + m * stride + kc;
    v8h lo = *(const v8h*)(base + half * 8);        // K = kc + half*8 .. +7
    v8h hi = *(const v8h*)(base + 16 + half * 8);   // K = kc + 16 + half*8 .. +7
    v16h r;
    #pragma unroll
    for (int j = 0; j < 8; ++j) { r[j] = lo[j]; r[8 + j] = hi[j]; }
    return r;
}

__device__ __forceinline__ v16h load_b_frag(const _Float16* __restrict__ Wf, int K,
                                            int ntile, int kc, int lane) {
    int half = lane >> 4;
    int nn   = (lane & 15) + ntile * 16;
    const _Float16* base = Wf + (size_t)nn * K + kc;
    v8h lo = *(const v8h*)(base + half * 8);        // K = kc + half*8 .. +7
    v8h hi = *(const v8h*)(base + 16 + half * 8);   // K = kc + 16 + half*8 .. +7
    v16h r;
    #pragma unroll
    for (int j = 0; j < 8; ++j) { r[j] = lo[j]; r[8 + j] = hi[j]; }
    return r;
}

__device__ __forceinline__ void store_relu_tile(_Float16* out, int CO, int mtile, int ntile,
                                                const v8f& acc, const float* __restrict__ bb,
                                                int lane) {
    int half = lane >> 4;
    int nn   = ntile * 16 + (lane & 15);
    float bias = bb[nn];
    #pragma unroll
    for (int r = 0; r < 8; ++r) {
        int row = mtile * 16 + half * 8 + r;
        float vv = acc[r] + bias;
        vv = vv > 0.f ? vv : 0.f;
        out[row * CO + nn] = (_Float16)vv;
    }
}

#define WMMA_F16(a, b, c) \
    __builtin_amdgcn_wmma_f32_16x16x32_f16(false, (a), false, (b), (short)0, (c), false, false)

// ---------------- kernel 2: MLP + max-pool (1 block = 1 (b,s), 4 waves) ----------------
__global__ void mlp_kernel(const _Float16* __restrict__ featG,
                           const _Float16* __restrict__ Wf0, const _Float16* __restrict__ Wf1,
                           const _Float16* __restrict__ Wf2,
                           const float* __restrict__ bb0, const float* __restrict__ bb1,
                           const float* __restrict__ bb2,
                           float* __restrict__ out) {
    __shared__ alignas(16) _Float16 bufIn[32 * 32];   // layer0 input, K padded to 32
    __shared__ alignas(16) _Float16 buf0[32 * 64];    // layer0 output
    __shared__ alignas(16) _Float16 buf1[32 * 128];   // layer1 output

    int bs   = blockIdx.x;
    int tid  = threadIdx.x;
    int wave = tid >> 5;
    int lane = tid & 31;

    // stage features: thread owns row=tid>>2, 8-col block cb=tid&3 (no overlap -> 1 barrier)
    {
        int row = tid >> 2, cb = tid & 3;
        _Float16* dst = bufIn + row * 32 + cb * 8;
        if (cb == 0) {
            const _Float16* f = featG + ((size_t)bs * NSS + row) * 8;
            #pragma unroll
            for (int j = 0; j < 8; ++j) dst[j] = f[j];
        } else {
            #pragma unroll
            for (int j = 0; j < 8; ++j) dst[j] = (_Float16)0.f;
        }
    }
    __syncthreads();

    // ---- layer 0: 32(pad) -> 64, 1 K-chunk, 4 N-tiles x 2 M-tiles = 8 jobs
    for (int job = wave; job < 8; job += 4) {
        int mtile = job & 1, ntile = job >> 1;
        v8f acc = {};
        v16h a = load_a_frag(bufIn, 32, mtile, 0, lane);
        v16h w = load_b_frag(Wf0, 32, ntile, 0, lane);
        acc = WMMA_F16(a, w, acc);
        store_relu_tile(buf0, 64, mtile, ntile, acc, bb0, lane);
    }
    __syncthreads();

    // ---- layer 1: 64 -> 128, 2 K-chunks, 8 N-tiles x 2 M-tiles = 16 jobs
    for (int job = wave; job < 16; job += 4) {
        int mtile = job & 1, ntile = job >> 1;
        v8f acc = {};
        #pragma unroll
        for (int kc = 0; kc < 64; kc += 32) {
            v16h a = load_a_frag(buf0, 64, mtile, kc, lane);
            v16h w = load_b_frag(Wf1, 64, ntile, kc, lane);
            acc = WMMA_F16(a, w, acc);
        }
        store_relu_tile(buf1, 128, mtile, ntile, acc, bb1, lane);
    }
    __syncthreads();

    // ---- layer 2: 128 -> 384, 4 K-chunks, 24 N-tiles; fuse ReLU + max over 32 rows
    for (int ntile = wave; ntile < 24; ntile += 4) {
        int nn = ntile * 16 + (lane & 15);
        float bias = bb2[nn];
        float colmax = 0.f;  // ReLU floor
        #pragma unroll
        for (int mtile = 0; mtile < 2; ++mtile) {
            v8f acc = {};
            #pragma unroll
            for (int kc = 0; kc < 128; kc += 32) {
                v16h a = load_a_frag(buf1, 128, mtile, kc, lane);
                v16h w = load_b_frag(Wf2, 128, ntile, kc, lane);
                acc = WMMA_F16(a, w, acc);
            }
            #pragma unroll
            for (int r = 0; r < 8; ++r) colmax = fmaxf(colmax, acc[r] + bias);
        }
        // merge row-halves held by lane and lane^16 (same column n)
        colmax = fmaxf(colmax, __shfl_xor(colmax, 16, 32));
        if (lane < 16) out[(size_t)bs * 384 + nn] = colmax;
    }
}

// ---------------- launcher ----------------
extern "C" void kernel_launch(void* const* d_in, const int* in_sizes, int n_in,
                              void* d_out, int out_size, void* d_ws, size_t ws_size,
                              hipStream_t stream) {
    const float* xyz     = (const float*)d_in[0];
    const float* centers = (const float*)d_in[1];
    const float* cns     = (const float*)d_in[2];
    const float* pts     = (const float*)d_in[3];
    const float* nori    = (const float*)d_in[4];
    const int*   idx     = (const int*)d_in[5];

    const float* W[3]  = { (const float*)d_in[6],  (const float*)d_in[12], (const float*)d_in[18] };
    const float* bv[3] = { (const float*)d_in[7],  (const float*)d_in[13], (const float*)d_in[19] };
    const float* g[3]  = { (const float*)d_in[8],  (const float*)d_in[14], (const float*)d_in[20] };
    const float* be[3] = { (const float*)d_in[9],  (const float*)d_in[15], (const float*)d_in[21] };
    const float* mm[3] = { (const float*)d_in[10], (const float*)d_in[16], (const float*)d_in[22] };
    const float* vv[3] = { (const float*)d_in[11], (const float*)d_in[17], (const float*)d_in[23] };

    char* ws = (char*)d_ws;
    _Float16* Wf0 = (_Float16*)(ws + WF0_OFF);
    _Float16* Wf1 = (_Float16*)(ws + WF1_OFF);
    _Float16* Wf2 = (_Float16*)(ws + WF2_OFF);
    float* bb0 = (float*)(ws + BB0_OFF);
    float* bb1 = (float*)(ws + BB1_OFF);
    float* bb2 = (float*)(ws + BB2_OFF);
    _Float16* featG = (_Float16*)(ws + FEAT_OFF);

    prep_layer<<<1, 128, 0, stream>>>(W[0], bv[0], g[0], be[0], mm[0], vv[0], 8,   64,  32,  Wf0, bb0);
    prep_layer<<<1, 128, 0, stream>>>(W[1], bv[1], g[1], be[1], mm[1], vv[1], 64,  128, 64,  Wf1, bb1);
    prep_layer<<<3, 128, 0, stream>>>(W[2], bv[2], g[2], be[2], mm[2], vv[2], 128, 384, 128, Wf2, bb2);

    geom_kernel<<<(Bb * Ss) / 4, 128, 0, stream>>>(xyz, centers, cns, pts, nori, idx, featG);

    mlp_kernel<<<Bb * Ss, 128, 0, stream>>>(featG, Wf0, Wf1, Wf2, bb0, bb1, bb2, (float*)d_out);
}
